// MultiHeadAttention_77833397338736
// MI455X (gfx1250) — matrix operand order repaired and checked by reading, not compiled, not run on hardware
//
#include <hip/hip_runtime.h>

// ---------------------------------------------------------------------------
// MI455X (gfx1250) multi-head attention: bf16 WMMA GEMMs + flash attention.
// Wave32; v_wmma_f32_16x16x32_bf16 with f32 accumulation.
// - All WMMA operands K-major: every fragment gather is 2x b128 loads.
// - Ping-pong LDS tiles + register prefetch: ONE barrier per inner tile.
// - Fused QKV projection: single pass over x produces Q, K and V^T.
// - Attention: S^T = K x Q^T and O^T = V^T x P^T so softmax stats and
//   rescaling are per-lane (2 shuffles per 32-key block).
// ---------------------------------------------------------------------------

typedef __attribute__((ext_vector_type(16))) __bf16 v16bf;
typedef __attribute__((ext_vector_type(8)))  float  v8f;

#define BATCH   16
#define SEQ_N   1024
#define EMB     768
#define H_HEADS 12
#define DHEAD   64

static __device__ __forceinline__ __bf16 f2bf(float f) { return (__bf16)f; }

union V16U { v16bf v; uint4 q[2]; };
union V8U  { uint4 q; __bf16 e[8]; };

// Load a 16-bf16 WMMA fragment from two 16-byte-aligned 8-bf16 chunks.
// Vector elements 0..7 -> VGPRs 0..3 (q[0]); elements 8..15 -> VGPRs 4..7 (q[1]).
static __device__ __forceinline__ v16bf load_frag16(const __bf16* p0, const __bf16* p1) {
    V16U u;
    u.q[0] = *reinterpret_cast<const uint4*>(p0);
    u.q[1] = *reinterpret_cast<const uint4*>(p1);
    return u.v;
}

// ------------------------------ fp32 -> bf16 -------------------------------
__global__ __launch_bounds__(256) void cvt_f32_to_bf16(
    const float* __restrict__ in, __bf16* __restrict__ out, int n)
{
    int i = blockIdx.x * blockDim.x + threadIdx.x;
    int stride = gridDim.x * blockDim.x;
    for (; i < n; i += stride) out[i] = (__bf16)in[i];
}

// ---------------- fp32 [rows][cols] -> bf16 transposed [cols][rows] --------
__global__ __launch_bounds__(256) void transpose_f32_to_bf16(
    const float* __restrict__ in, __bf16* __restrict__ out, int rows, int cols)
{
    __shared__ float tile[32][33];
    const int c0 = blockIdx.x * 32, r0 = blockIdx.y * 32;
    const int tx = threadIdx.x & 31, ty = threadIdx.x >> 5; // ty: 0..7
#pragma unroll
    for (int i = 0; i < 32; i += 8)
        tile[ty + i][tx] = in[(size_t)(r0 + ty + i) * cols + (c0 + tx)];
    __syncthreads();
#pragma unroll
    for (int i = 0; i < 32; i += 8)
        out[(size_t)(c0 + ty + i) * rows + (r0 + tx)] = (__bf16)tile[tx][ty + i];
}

// ------------------------------ tiled WMMA GEMM ----------------------------
// C[M,N] = A[M,K] @ Bt[N,K]^T + bias, ping-pong LDS, one barrier per K-tile.
// MODE 0: fused QKV projection (N = 2304 = 1536 QK cols + 768 V cols)
//         cols [0,1536): Q (scaled 1/8) / K in [B,H,N,64] bf16 (bias = b_qk)
//         cols [1536,2304): V transposed [B,H,64,N] bf16      (bias2 = b_v)
// MODE 2: output projection -> f32 dense [M,N] (bias = b_proj)
// Block tile 128x64, 8 waves (4x2), wave tile 32x32, K-step 64 (2 WMMA steps).
template <int MODE>
__global__ __launch_bounds__(256) void gemm_bf16_wmma(
    const __bf16* __restrict__ A, const __bf16* __restrict__ Bt,
    const float* __restrict__ bias, const float* __restrict__ bias2,
    int M, int N, int K,
    __bf16* __restrict__ outQ, __bf16* __restrict__ outK,
    __bf16* __restrict__ outV, float* __restrict__ outF)
{
    __shared__ __attribute__((aligned(16))) __bf16 As [2][128 * 72]; // 128 rows x 64 K
    __shared__ __attribute__((aligned(16))) __bf16 Bts[2][ 64 * 72]; //  64 cols x 64 K

    const int tid  = threadIdx.x;
    const int lane = tid & 31;
    const int wave = tid >> 5;
    const int hl   = lane >> 4;     // wave half (0: lanes 0-15, 1: lanes 16-31)
    const int ln   = lane & 15;
    const int wm   = wave >> 1;     // 0..3 -> 32-row slab
    const int wn   = wave & 1;      // 0..1 -> 32-col slab
    const int tile_m = blockIdx.x * 128;
    const int tile_n = blockIdx.y * 64;

    v8f acc[2][2] = {};

    const int arow = tid >> 1, aseg = tid & 1;  // each thread: 32 bf16 of A tile
    const int bcol = tid >> 2, bseg = tid & 3;  // each thread: 16 bf16 of Bt tile
    const __bf16* gA = A  + (size_t)(tile_m + arow) * K + aseg * 32;
    const __bf16* gB = Bt + (size_t)(tile_n + bcol) * K + bseg * 16;

    uint4 ra0, ra1, ra2, ra3, rb0, rb1;
    auto load_regs = [&](int kt) {
        const uint4* pa = reinterpret_cast<const uint4*>(gA + kt);
        ra0 = pa[0]; ra1 = pa[1]; ra2 = pa[2]; ra3 = pa[3];
        const uint4* pb = reinterpret_cast<const uint4*>(gB + kt);
        rb0 = pb[0]; rb1 = pb[1];
    };
    auto store_tile = [&](int p) {
        uint4* la = reinterpret_cast<uint4*>(&As[p][arow * 72 + aseg * 32]);
        la[0] = ra0; la[1] = ra1; la[2] = ra2; la[3] = ra3;
        uint4* lb = reinterpret_cast<uint4*>(&Bts[p][bcol * 72 + bseg * 16]);
        lb[0] = rb0; lb[1] = rb1;
    };
    auto compute_tile = [&](int p) {
#pragma unroll
        for (int kk = 0; kk < 64; kk += 32) {
            v16bf af[2], bfr[2];
#pragma unroll
            for (int s = 0; s < 2; ++s) {
                const int row_l = wm * 32 + s * 16 + ln;
                af[s] = load_frag16(&As[p][row_l * 72 + kk + hl * 8],
                                    &As[p][row_l * 72 + kk + 16 + hl * 8]);
                const int col_l = wn * 32 + s * 16 + ln;
                bfr[s] = load_frag16(&Bts[p][col_l * 72 + kk + hl * 8],
                                     &Bts[p][col_l * 72 + kk + 16 + hl * 8]);
            }
#pragma unroll
            for (int mi = 0; mi < 2; ++mi)
#pragma unroll
                for (int ni = 0; ni < 2; ++ni)
                    acc[mi][ni] = __builtin_amdgcn_wmma_f32_16x16x32_bf16(
                        false, af[mi], false, bfr[ni], (short)0, acc[mi][ni], false, false);
        }
    };

    load_regs(0);
    for (int kt = 0; kt < K; kt += 128) {      // K % 128 == 0
        store_tile(0);
        __syncthreads();
        load_regs(kt + 64);                    // kt+64 < K always
        compute_tile(0);
        store_tile(1);
        __syncthreads();
        if (kt + 128 < K) load_regs(kt + 128);
        compute_tile(1);
    }

    // ---- epilogue: C layout lane (hl, ln) VGPR v -> (M = v + 8*hl, N = ln) ----
#pragma unroll
    for (int mi = 0; mi < 2; ++mi)
#pragma unroll
        for (int ni = 0; ni < 2; ++ni)
#pragma unroll
            for (int v = 0; v < 8; ++v) {
                const int row = tile_m + wm * 32 + mi * 16 + v + 8 * hl;
                const int col = tile_n + wn * 32 + ni * 16 + ln;
                if (MODE == 0) {
                    const int bb = row >> 10, nn = row & 1023;
                    if (col < 1536) {          // uniform per 64-col tile
                        const float val = acc[mi][ni][v] + bias[col];
                        const int h   = col >> 7;
                        const int rem = col & 127;
                        const int dd  = rem >> 1;
                        const size_t idx =
                            (((size_t)bb * H_HEADS + h) * SEQ_N + nn) * DHEAD + dd;
                        if (rem & 1) outK[idx] = f2bf(val);
                        else         outQ[idx] = f2bf(val * 0.125f); // 1/sqrt(64)
                    } else {
                        const int c2 = col - 1536;
                        const float val = acc[mi][ni][v] + bias2[c2];
                        const int h  = c2 >> 6;
                        const int dd = c2 & 63;
                        const size_t idx = // transposed: [B,H,64,N]
                            (((size_t)bb * H_HEADS + h) * DHEAD + dd) * SEQ_N + nn;
                        outV[idx] = f2bf(val);
                    }
                } else {
                    outF[(size_t)row * N + col] = acc[mi][ni][v] + bias[col];
                }
            }
}

// ------------------------------ flash attention ----------------------------
// Q,K: bf16 [B,H,N,64] (Q pre-scaled by 1/8). Vt: bf16 [B,H,64,N].
// Out: bf16 [B,N,768]. 8 waves/block share one (b,h); block covers 128
// queries (16 per wave). Ping-pong K/V tiles: one barrier per 32-key block.
__global__ __launch_bounds__(256) void attn_fmha(
    const __bf16* __restrict__ Q, const __bf16* __restrict__ Kt,
    const __bf16* __restrict__ Vt, __bf16* __restrict__ O)
{
    __shared__ __attribute__((aligned(16))) __bf16 Ks[2][32 * 72];  // 32 keys x 64 dims
    __shared__ __attribute__((aligned(16))) __bf16 Vs[2][64 * 40];  // 64 dv  x 32 keys
    __shared__ __attribute__((aligned(16))) __bf16 Pbuf[8][16][32]; // per-wave P^T

    const int tid  = threadIdx.x;
    const int lane = tid & 31;
    const int wave = tid >> 5;
    const int hl   = lane >> 4;
    const int ln   = lane & 15;
    const int bh   = blockIdx.x >> 3;   // 8 query-blocks of 128 per (b,h)
    const int qblk = blockIdx.x & 7;
    const int b = bh / H_HEADS, h = bh % H_HEADS;
    const int q0 = qblk * 128 + wave * 16;

    const size_t base = ((size_t)b * H_HEADS + h) * SEQ_N * DHEAD;
    const __bf16* Qp  = Q  + base;
    const __bf16* Kp  = Kt + base;
    const __bf16* Vtp = Vt + base;      // [64][1024] per (b,h)

    // Q^T fragments (B layout): col = query ln, reduction = head dims
    const int qrow = q0 + ln;
    const v16bf qb0 = load_frag16(&Qp[qrow * DHEAD + hl * 8],
                                  &Qp[qrow * DHEAD + 16 + hl * 8]);
    const v16bf qb1 = load_frag16(&Qp[qrow * DHEAD + 32 + hl * 8],
                                  &Qp[qrow * DHEAD + 48 + hl * 8]);

    // cooperative K/V tile staging assignment (1 uint4 of each per thread)
    const int krow = tid >> 3, kseg = tid & 7;   // K: 32 rows x 64 dims
    const int vrow = tid >> 2, vseg = tid & 3;   // V: 64 rows x 32 keys
    const __bf16* gK = Kp  + krow * DHEAD + kseg * 8;
    const __bf16* gV = Vtp + (size_t)vrow * SEQ_N + vseg * 8;

    uint4 rk, rv;
    auto load_regs = [&](int kb) {
        rk = *reinterpret_cast<const uint4*>(gK + kb * DHEAD);
        rv = *reinterpret_cast<const uint4*>(gV + kb);
    };
    auto store_tile = [&](int p) {
        *reinterpret_cast<uint4*>(&Ks[p][krow * 72 + kseg * 8]) = rk;
        *reinterpret_cast<uint4*>(&Vs[p][vrow * 40 + vseg * 8]) = rv;
    };

    // per-lane online softmax state for query q0+ln (replicated across hl)
    float m_run = -1e30f, l_run = 0.0f;
    v8f oacc[4] = {};   // O^T: frag t rows dv = 16t + v + 8*hl, col = query ln

    auto process_block = [&](int p) {
        // K A-fragments from LDS: rows = keys, reduction = head dims
        const v16bf kA0 = load_frag16(&Ks[p][ln * 72 + hl * 8],
                                      &Ks[p][ln * 72 + 16 + hl * 8]);
        const v16bf kA1 = load_frag16(&Ks[p][ln * 72 + 32 + hl * 8],
                                      &Ks[p][ln * 72 + 48 + hl * 8]);
        const v16bf kB0 = load_frag16(&Ks[p][(ln + 16) * 72 + hl * 8],
                                      &Ks[p][(ln + 16) * 72 + 16 + hl * 8]);
        const v16bf kB1 = load_frag16(&Ks[p][(ln + 16) * 72 + 32 + hl * 8],
                                      &Ks[p][(ln + 16) * 72 + 48 + hl * 8]);

        // S^T tiles: [keys x queries]; lane holds keys v+8*hl (+16) @ query ln
        v8f z = {};
        v8f sT0 = __builtin_amdgcn_wmma_f32_16x16x32_bf16(false, kA0, false, qb0, (short)0, z,   false, false);
        sT0     = __builtin_amdgcn_wmma_f32_16x16x32_bf16(false, kA1, false, qb1, (short)0, sT0, false, false);
        v8f sT1 = __builtin_amdgcn_wmma_f32_16x16x32_bf16(false, kB0, false, qb0, (short)0, z,   false, false);
        sT1     = __builtin_amdgcn_wmma_f32_16x16x32_bf16(false, kB1, false, qb1, (short)0, sT1, false, false);

        // per-lane online softmax over this 32-key block
        float mb = fmaxf(sT0[0], sT1[0]);
#pragma unroll
        for (int v = 1; v < 8; ++v) mb = fmaxf(mb, fmaxf(sT0[v], sT1[v]));
        mb = fmaxf(mb, __shfl_xor(mb, 16, 32));        // combine wave halves
        const float mn = fmaxf(m_run, mb);
        const float sc = __expf(m_run - mn);
        float rs = 0.0f;
        V8U u0, u1;
#pragma unroll
        for (int v = 0; v < 8; ++v) {
            const float p0 = __expf(sT0[v] - mn);
            const float p1 = __expf(sT1[v] - mn);
            rs += p0 + p1;
            u0.e[v] = f2bf(p0);
            u1.e[v] = f2bf(p1);
        }
        rs += __shfl_xor(rs, 16, 32);
        l_run = l_run * sc + rs;
        m_run = mn;
#pragma unroll
        for (int t = 0; t < 4; ++t)
#pragma unroll
            for (int v = 0; v < 8; ++v) oacc[t][v] *= sc;

        // P^T -> per-wave LDS [query][key]; lane's 8 values are 8 consecutive
        // keys -> packed uint4 stores. Same-wave LDS is in-order; only block
        // compiler reordering.
        *reinterpret_cast<uint4*>(&Pbuf[wave][ln][8 * hl])      = u0.q;
        *reinterpret_cast<uint4*>(&Pbuf[wave][ln][16 + 8 * hl]) = u1.q;
        asm volatile("" ::: "memory");

        // P^T B-fragment: col = query ln, reduction = keys
        const v16bf pT = load_frag16(&Pbuf[wave][ln][hl * 8],
                                     &Pbuf[wave][ln][16 + hl * 8]);
        // V^T A-fragments: rows = dv, reduction = keys; O^T += V^T x P^T
#pragma unroll
        for (int t = 0; t < 4; ++t) {
            const int dvr = t * 16 + ln;
            const v16bf vT = load_frag16(&Vs[p][dvr * 40 + hl * 8],
                                         &Vs[p][dvr * 40 + 16 + hl * 8]);
            oacc[t] = __builtin_amdgcn_wmma_f32_16x16x32_bf16(
                false, vT, false, pT, (short)0, oacc[t], false, false);
        }
    };

    load_regs(0);
    for (int kb = 0; kb < SEQ_N; kb += 64) {   // SEQ_N % 64 == 0
        store_tile(0);
        __syncthreads();
        load_regs(kb + 32);                    // kb+32 < SEQ_N always
        process_block(0);
        store_tile(1);
        __syncthreads();
        if (kb + 64 < SEQ_N) load_regs(kb + 64);
        process_block(1);
    }

    // finalize: per-lane 1/l, packed stores (8 consecutive dv per lane)
    const float inv_l = 1.0f / l_run;
    const size_t orow = ((size_t)b * SEQ_N + q0 + ln) * EMB + h * DHEAD;
#pragma unroll
    for (int t = 0; t < 4; ++t) {
        V8U u;
#pragma unroll
        for (int v = 0; v < 8; ++v) u.e[v] = f2bf(oacc[t][v] * inv_l);
        *reinterpret_cast<uint4*>(&O[orow + t * 16 + 8 * hl]) = u.q;
    }
}

// ------------------------------ launch -------------------------------------
extern "C" void kernel_launch(void* const* d_in, const int* in_sizes, int n_in,
                              void* d_out, int out_size, void* d_ws, size_t ws_size,
                              hipStream_t stream)
{
    const float* x      = (const float*)d_in[0];
    const float* W_qk   = (const float*)d_in[1];
    const float* b_qk   = (const float*)d_in[2];
    const float* W_v    = (const float*)d_in[3];
    const float* b_v    = (const float*)d_in[4];
    const float* W_proj = (const float*)d_in[5];
    const float* b_proj = (const float*)d_in[6];
    float* out = (float*)d_out;
    (void)in_sizes; (void)n_in; (void)out_size; (void)ws_size;

    char* ws = (char*)d_ws;
    size_t off = 0;
    auto carve = [&](size_t bytes) -> char* {
        char* p = ws + off;
        off += (bytes + 255) & ~(size_t)255;
        return p;
    };
    const int NX = BATCH * SEQ_N * EMB;   // 12,582,912

    __bf16* xb    = (__bf16*)carve((size_t)NX * 2);
    __bf16* wqkvt = (__bf16*)carve((size_t)2304 * EMB * 2); // [2304][768] fused
    __bf16* wpt   = (__bf16*)carve((size_t)EMB * EMB * 2);  // [768][768]
    __bf16* Qb    = (__bf16*)carve((size_t)NX * 2);         // [B,H,N,64]
    __bf16* Kb    = (__bf16*)carve((size_t)NX * 2);         // [B,H,N,64]
    __bf16* Vtb   = (__bf16*)carve((size_t)NX * 2);         // [B,H,64,N]
    __bf16* Ab    = (__bf16*)carve((size_t)NX * 2);         // [B,N,768]

    cvt_f32_to_bf16<<<(NX + 1023) / 1024, 256, 0, stream>>>(x, xb, NX);
    // fused transposed weights: rows [0,1536) = W_qk^T, rows [1536,2304) = W_v^T
    transpose_f32_to_bf16<<<dim3(1536 / 32, EMB / 32), 256, 0, stream>>>(
        W_qk, wqkvt, EMB, 1536);
    transpose_f32_to_bf16<<<dim3(EMB / 32, EMB / 32), 256, 0, stream>>>(
        W_v, wqkvt + (size_t)1536 * EMB, EMB, EMB);
    transpose_f32_to_bf16<<<dim3(EMB / 32, EMB / 32), 256, 0, stream>>>(
        W_proj, wpt, EMB, EMB);

    const int M = BATCH * SEQ_N; // 16384
    dim3 gQKV(M / 128, 2304 / 64);
    gemm_bf16_wmma<0><<<gQKV, 256, 0, stream>>>(xb, wqkvt, b_qk, b_v,
                                                M, 2304, EMB,
                                                Qb, Kb, Vtb, nullptr);

    attn_fmha<<<BATCH * H_HEADS * (SEQ_N / 128), 256, 0, stream>>>(Qb, Kb, Vtb, Ab);

    dim3 gP(M / 128, EMB / 64);
    gemm_bf16_wmma<2><<<gP, 256, 0, stream>>>(Ab, wpt, b_proj, nullptr,
                                              M, EMB, EMB,
                                              nullptr, nullptr, nullptr, out);
}
